// MultiHeadAttention_47502338294120
// MI455X (gfx1250) — matrix operand order repaired and checked by reading, not compiled, not run on hardware
//
#include <hip/hip_runtime.h>
#include <hip/hip_bf16.h>

// ---------------------------------------------------------------------------
// GQA attention forward, MI455X (gfx1250), wave32, bf16 WMMA w/ f32 accum.
// B=1, S=4096, H=2048, NH=16, NKV=4, HD=128, GROUPS=4.
// GEMMs: LDS-staged, double-buffered B tile filled via async global->LDS
// (ASYNCcnt / s_wait_asynccnt), 32x128 wave tiles, v_wmma_f32_16x16x32_bf16.
// ---------------------------------------------------------------------------

#define S_LEN 4096
#define HID   2048
#define NH_   16
#define NKV_  4
#define HD_   128

#define USE_ASYNC_LDS 1

typedef __bf16         v16bf  __attribute__((ext_vector_type(16)));
typedef float          v8f    __attribute__((ext_vector_type(8)));
typedef unsigned int   u32x4  __attribute__((ext_vector_type(4)));
typedef float          f32x4  __attribute__((ext_vector_type(4)));
typedef unsigned short u16x4  __attribute__((ext_vector_type(4)));

union BfOp { v16bf v; u32x4 u[2]; };

static __device__ __forceinline__ unsigned short bf16bits(float f) {
  unsigned u = __builtin_bit_cast(unsigned, f);
  u += 0x7FFFu + ((u >> 16) & 1u);          // round-to-nearest-even
  return (unsigned short)(u >> 16);
}
static __device__ __forceinline__ __bf16 tobf(float f) {
  unsigned short h = bf16bits(f);
  return __builtin_bit_cast(__bf16, h);
}

static __device__ __forceinline__ v8f wmma_bf16(v16bf a, v16bf b, v8f c) {
  // D(16x16,f32) = A(16x32,bf16) x B(32x16,bf16) + C
  return __builtin_amdgcn_wmma_f32_16x16x32_bf16(
      false, a, false, b, (short)0, c, false, false);
}

// 16-byte per-lane async copy global -> LDS (GLOBAL_LOAD_ASYNC_TO_LDS_B128)
static __device__ __forceinline__ void async_copy16(const __bf16* g, __bf16* l) {
#if USE_ASYNC_LDS
  unsigned           lo = (unsigned)(uintptr_t)l;            // LDS byte offset
  unsigned long long ga = (unsigned long long)(uintptr_t)g;  // global address
  asm volatile("global_load_async_to_lds_b128 %0, %1, off"
               :: "v"(lo), "v"(ga) : "memory");
#else
  *reinterpret_cast<u32x4*>(l) = *reinterpret_cast<const u32x4*>(g);
#endif
}
static __device__ __forceinline__ void async_wait0() {
#if USE_ASYNC_LDS
  asm volatile("s_wait_asynccnt 0" ::: "memory");
#endif
}

// ---------------------------------------------------------------------------
// Stage 0a: f32 -> bf16 elementwise (vectorized x4)
// ---------------------------------------------------------------------------
__global__ __launch_bounds__(256)
void conv_f32_bf16(const float* __restrict__ in, __bf16* __restrict__ out, int n4) {
  int i = blockIdx.x * blockDim.x + threadIdx.x;
  if (i >= n4) return;
  f32x4 f = ((const f32x4*)in)[i];
  u16x4 h;
  h.x = bf16bits(f.x); h.y = bf16bits(f.y);
  h.z = bf16bits(f.z); h.w = bf16bits(f.w);
  ((u16x4*)out)[i] = h;
}

// ---------------------------------------------------------------------------
// Stage 0b: W (KxN f32, row-major) -> W^T (NxK bf16, row-major), LDS-tiled
// ---------------------------------------------------------------------------
__global__ __launch_bounds__(256)
void transpose_conv_bf16(const float* __restrict__ in, __bf16* __restrict__ out,
                         int K, int N) {
  __shared__ __bf16 tile[32][33];
  int k0 = blockIdx.y * 32, n0 = blockIdx.x * 32;
  #pragma unroll
  for (int r = threadIdx.y; r < 32; r += 8)
    tile[r][threadIdx.x] = tobf(in[(size_t)(k0 + r) * N + n0 + threadIdx.x]);
  __syncthreads();
  #pragma unroll
  for (int r = threadIdx.y; r < 32; r += 8)
    out[(size_t)(n0 + r) * K + k0 + threadIdx.x] = tile[threadIdx.x][r];
}

// ---------------------------------------------------------------------------
// Stage 1/4: bf16 WMMA GEMM.  C[M,N] = A[M,K] x Bt[N,K]^T
// Block: 256 thr = 8 waves, tile 256(M) x 128(N); wave tile 32(M) x 128(N).
// B tile (128 x 32 bf16) staged in LDS, double-buffered, async-filled.
// LDS rows padded to 40 elems (80B = 20 dwords, coprime vs 64 banks).
// MODE 0: out bf16 -> [head][s][128]      (Q, head = n>>7)
// MODE 1: out bf16 -> [kvh ][s][128]      (K)
// MODE 2: out bf16 -> [kvh ][128][S]      (V transposed)
// MODE 3: out f32  -> row-major [M][N]    (final projection)
// ---------------------------------------------------------------------------
template <int MODE>
__global__ __launch_bounds__(256)
void gemm_bf16_wmma(const __bf16* __restrict__ A, const __bf16* __restrict__ Bt,
                    void* __restrict__ Out, int M, int N, int K) {
  __shared__ __attribute__((aligned(16))) __bf16 ldsb[2][128][40];

  const int tid  = threadIdx.x;
  const int lane = tid & 31;
  const int wave = tid >> 5;
  const int m0   = blockIdx.y * 256 + wave * 32;
  const int n0   = blockIdx.x * 128;
  const int half = lane & 15;
  const bool hi  = lane >= 16;

  // B-tile fill assignment: row fr & fr+64, 16B chunk fc
  const int fr = tid >> 2;          // 0..63
  const int fc = (tid & 3) * 8;     // elem offset 0/8/16/24

  // A operands: lane<16 holds K 0..7 & 16..23; lane>=16 holds 8..15 & 24..31
  const __bf16* a0row = A + (size_t)(m0 + half) * K + (hi ? 8 : 0);
  const __bf16* a1row = a0row + (size_t)16 * K;
  const int bko = hi ? 8 : 0;       // B operand: lanes 0-15 K 0..7 & 16..23, 16-31 K 8..15 & 24..31

  v8f acc0[8] = {}, acc1[8] = {};

  // prologue: fill buffer 0
  {
    const __bf16* g = Bt + (size_t)(n0 + fr) * K + fc;
    async_copy16(g,                 &ldsb[0][fr][fc]);
    async_copy16(g + (size_t)64 * K, &ldsb[0][fr + 64][fc]);
  }
  async_wait0();
  __syncthreads();

  int buf = 0;
  for (int k = 0; k < K; k += 32) {
    // issue async fill of the other buffer for the next k-step
    if (k + 32 < K) {
      const __bf16* g = Bt + (size_t)(n0 + fr) * K + (k + 32) + fc;
      async_copy16(g,                 &ldsb[buf ^ 1][fr][fc]);
      async_copy16(g + (size_t)64 * K, &ldsb[buf ^ 1][fr + 64][fc]);
    }
    __builtin_prefetch((const void*)(a0row + k + 128), 0, 0);

    BfOp a0, a1;
    a0.u[0] = *reinterpret_cast<const u32x4*>(a0row + k);
    a0.u[1] = *reinterpret_cast<const u32x4*>(a0row + k + 16);
    a1.u[0] = *reinterpret_cast<const u32x4*>(a1row + k);
    a1.u[1] = *reinterpret_cast<const u32x4*>(a1row + k + 16);

    #pragma unroll
    for (int t = 0; t < 8; ++t) {
      BfOp b;
      b.u[0] = *reinterpret_cast<const u32x4*>(&ldsb[buf][t * 16 + half][bko]);
      b.u[1] = *reinterpret_cast<const u32x4*>(&ldsb[buf][t * 16 + half][bko + 16]);
      acc0[t] = wmma_bf16(a0.v, b.v, acc0[t]);
      acc1[t] = wmma_bf16(a1.v, b.v, acc1[t]);
    }

    async_wait0();       // our wave's async fills for next buffer complete
    __syncthreads();     // everyone done reading buf & done filling buf^1
    buf ^= 1;
  }

  #pragma unroll
  for (int half_m = 0; half_m < 2; ++half_m) {
    v8f* acc = half_m ? acc1 : acc0;
    #pragma unroll
    for (int t = 0; t < 8; ++t) {
      #pragma unroll
      for (int j = 0; j < 8; ++j) {
        const int row = m0 + half_m * 16 + (hi ? j + 8 : j);
        const int n   = n0 + t * 16 + half;
        const float v = acc[t][j];
        if (MODE == 3) {
          ((float*)Out)[(size_t)row * N + n] = v;
        } else if (MODE == 2) {
          const int h = n >> 7, hd = n & 127;
          ((__bf16*)Out)[((size_t)h * HD_ + hd) * S_LEN + row] = tobf(v);
        } else {
          const int h = n >> 7, hd = n & 127;
          ((__bf16*)Out)[((size_t)h * S_LEN + row) * HD_ + hd] = tobf(v);
        }
      }
    }
  }
}

// ---------------------------------------------------------------------------
// Stage 2: causal flash attention (GQA).  One wave = 16 query rows x HD=128.
// Q: [NH][S][128] bf16, K: [NKV][S][128] bf16, Vt: [NKV][128][S] bf16.
// Ctx out: [S][H] bf16 (head-concatenated).
// ---------------------------------------------------------------------------
__global__ __launch_bounds__(256)
void attn_fwd(const __bf16* __restrict__ Q, const __bf16* __restrict__ Kb,
              const __bf16* __restrict__ Vt, __bf16* __restrict__ Ctx) {
  __shared__ __attribute__((aligned(16))) __bf16 plds[8][16][32];

  const int head = blockIdx.x;
  const int kvh  = head >> 2;                 // GROUPS = 4
  const int lane = threadIdx.x & 31;
  const int wave = threadIdx.x >> 5;
  const int q0   = blockIdx.y * 128 + wave * 16;
  const int half = lane & 15;
  const bool hi  = lane >= 16;

  const __bf16* qrow  = Q  + ((size_t)head * S_LEN + q0 + half) * HD_ + (hi ? 8 : 0);
  const __bf16* kbase = Kb + (size_t)kvh * S_LEN * HD_;
  const __bf16* vbase = Vt + (size_t)kvh * HD_ * S_LEN;

  // Preload Q as 4 A-operands (d chunks of 32)
  v16bf aq[4];
  #pragma unroll
  for (int c = 0; c < 4; ++c) {
    BfOp t;
    t.u[0] = *reinterpret_cast<const u32x4*>(qrow + c * 32);
    t.u[1] = *reinterpret_cast<const u32x4*>(qrow + c * 32 + 16);
    aq[c] = t.v;
  }

  v8f acc[8] = {};
  float mrow[8], lrow[8];
  #pragma unroll
  for (int j = 0; j < 8; ++j) { mrow[j] = -1e30f; lrow[j] = 0.0f; }

  const float scale = 0.08838834764831845f;   // 1/sqrt(128)
  const int dof = hi ? 8 : 0;                 // B-operand K split
  const int kv_end = q0 + 16;                 // causal bound

  for (int kb = 0; kb < kv_end; kb += 32) {
    // ---- scores S = Q K^T : two 16x16 tiles (kv kb..kb+15, kb+16..kb+31)
    v8f s0 = {}, s1 = {};
    #pragma unroll
    for (int c = 0; c < 4; ++c) {
      const __bf16* kp0 = kbase + (size_t)(kb + half) * HD_ + c * 32 + dof;
      const __bf16* kp1 = kp0 + 16 * HD_;
      BfOp b0, b1;
      b0.u[0] = *reinterpret_cast<const u32x4*>(kp0);
      b0.u[1] = *reinterpret_cast<const u32x4*>(kp0 + 16);
      b1.u[0] = *reinterpret_cast<const u32x4*>(kp1);
      b1.u[1] = *reinterpret_cast<const u32x4*>(kp1 + 16);
      s0 = wmma_bf16(aq[c], b0.v, s0);
      s1 = wmma_bf16(aq[c], b1.v, s1);
    }

    // ---- online softmax (row = 16 lanes of one half-wave, per C-layout)
    #pragma unroll
    for (int j = 0; j < 8; ++j) {
      const int row = q0 + (hi ? j + 8 : j);
      const int c0 = kb + half, c1 = c0 + 16;
      float x0 = (c0 <= row) ? s0[j] * scale : -1e30f;
      float x1 = (c1 <= row) ? s1[j] * scale : -1e30f;
      float mx = fmaxf(x0, x1);
      #pragma unroll
      for (int off = 8; off >= 1; off >>= 1)
        mx = fmaxf(mx, __shfl_xor(mx, off, 32));
      const float mnew = fmaxf(mrow[j], mx);
      const float p0 = __expf(x0 - mnew);
      const float p1 = __expf(x1 - mnew);
      float ps = p0 + p1;
      #pragma unroll
      for (int off = 8; off >= 1; off >>= 1)
        ps += __shfl_xor(ps, off, 32);
      const float corr = __expf(mrow[j] - mnew);
      lrow[j] = lrow[j] * corr + ps;
      mrow[j] = mnew;
      #pragma unroll
      for (int t = 0; t < 8; ++t) acc[t][j] *= corr;
      // stash P (bf16) in score layout -> LDS, to be re-read in A layout
      plds[wave][hi ? j + 8 : j][half]      = tobf(p0);
      plds[wave][hi ? j + 8 : j][half + 16] = tobf(p1);
    }

    // wave-private LDS roundtrip: enforce store->load ordering
    asm volatile("s_wait_dscnt 0" ::: "memory");

    // ---- P as A-operand (16 q x 32 kv)
    BfOp pa;
    pa.u[0] = *reinterpret_cast<const u32x4*>(&plds[wave][half][hi ? 8 : 0]);
    pa.u[1] = *reinterpret_cast<const u32x4*>(&plds[wave][half][(hi ? 8 : 0) + 16]);

    // ---- ctx += P x V : 8 d-subtiles, B from Vt ([d][s], contiguous kv)
    #pragma unroll
    for (int t = 0; t < 8; ++t) {
      const __bf16* vp = vbase + (size_t)(t * 16 + half) * S_LEN + kb + dof;
      BfOp bv;
      bv.u[0] = *reinterpret_cast<const u32x4*>(vp);
      bv.u[1] = *reinterpret_cast<const u32x4*>(vp + 16);
      acc[t] = wmma_bf16(pa.v, bv.v, acc[t]);
    }
  }

  // ---- normalize and emit context (bf16, [S][H] head-concatenated)
  #pragma unroll
  for (int j = 0; j < 8; ++j) {
    const int row = q0 + (hi ? j + 8 : j);
    const float inv = 1.0f / lrow[j];
    #pragma unroll
    for (int t = 0; t < 8; ++t) {
      Ctx[(size_t)row * HID + head * HD_ + t * 16 + half] = tobf(acc[t][j] * inv);
    }
  }
}

// ---------------------------------------------------------------------------
// Host-side launch
// ---------------------------------------------------------------------------
extern "C" void kernel_launch(void* const* d_in, const int* in_sizes, int n_in,
                              void* d_out, int out_size, void* d_ws, size_t ws_size,
                              hipStream_t stream) {
  (void)in_sizes; (void)n_in; (void)out_size; (void)ws_size;
  const float* X  = (const float*)d_in[0];
  // d_in[1] = attention_mask: pure causal structure, computed inline instead.
  const float* Wq = (const float*)d_in[2];
  const float* Wk = (const float*)d_in[3];
  const float* Wv = (const float*)d_in[4];
  const float* Wo = (const float*)d_in[5];

  char* ws = (char*)d_ws;
  size_t off = 0;
  auto alloc = [&](size_t bytes) {
    char* p = ws + off;
    off += (bytes + 255) & ~(size_t)255;
    return p;
  };
  __bf16* Xb  = (__bf16*)alloc((size_t)S_LEN * HID * 2);          // 16 MB
  __bf16* Wqt = (__bf16*)alloc((size_t)HID * HID * 2);            //  8 MB
  __bf16* Wkt = (__bf16*)alloc((size_t)512 * HID * 2);            //  2 MB
  __bf16* Wvt = (__bf16*)alloc((size_t)512 * HID * 2);            //  2 MB
  __bf16* Wot = (__bf16*)alloc((size_t)HID * HID * 2);            //  8 MB
  __bf16* Qb  = (__bf16*)alloc((size_t)NH_ * S_LEN * HD_ * 2);    // 16 MB
  __bf16* Kb  = (__bf16*)alloc((size_t)NKV_ * S_LEN * HD_ * 2);   //  4 MB
  __bf16* Vtb = (__bf16*)alloc((size_t)NKV_ * HD_ * S_LEN * 2);   //  4 MB
  __bf16* Ctx = Xb;  // X-bf16 is dead once Q/K/V are built; reuse for context

  // Stage 0: conversions / weight transposes
  {
    int n4 = S_LEN * HID / 4;
    conv_f32_bf16<<<n4 / 256, 256, 0, stream>>>(X, Xb, n4);
    dim3 blk(32, 8);
    transpose_conv_bf16<<<dim3(HID / 32, HID / 32), blk, 0, stream>>>(Wq, Wqt, HID, HID);
    transpose_conv_bf16<<<dim3(512 / 32, HID / 32), blk, 0, stream>>>(Wk, Wkt, HID, 512);
    transpose_conv_bf16<<<dim3(512 / 32, HID / 32), blk, 0, stream>>>(Wv, Wvt, HID, 512);
    transpose_conv_bf16<<<dim3(HID / 32, HID / 32), blk, 0, stream>>>(Wo, Wot, HID, HID);
  }

  // Stage 1: projections (WMMA GEMMs, 256x128 block tiles)
  gemm_bf16_wmma<0><<<dim3(HID / 128, S_LEN / 256), 256, 0, stream>>>(
      Xb, Wqt, Qb, S_LEN, HID, HID);
  gemm_bf16_wmma<1><<<dim3(512 / 128, S_LEN / 256), 256, 0, stream>>>(
      Xb, Wkt, Kb, S_LEN, 512, HID);
  gemm_bf16_wmma<2><<<dim3(512 / 128, S_LEN / 256), 256, 0, stream>>>(
      Xb, Wvt, Vtb, S_LEN, 512, HID);

  // Stage 2: causal flash attention (16 heads x 32 q-blocks)
  attn_fwd<<<dim3(NH_, S_LEN / 128), 256, 0, stream>>>(Qb, Kb, Vtb, Ctx);

  // Stage 3: output projection, f32 result
  gemm_bf16_wmma<3><<<dim3(HID / 128, S_LEN / 256), 256, 0, stream>>>(
      Ctx, Wot, d_out, S_LEN, HID, HID);
}
